// QNetwork_14620068675624
// MI455X (gfx1250) — matrix operand order repaired and checked by reading, not compiled, not run on hardware
//
#include <hip/hip_runtime.h>

typedef __attribute__((ext_vector_type(16))) _Float16 v16h;
typedef __attribute__((ext_vector_type(8)))  _Float16 v8h;
typedef __attribute__((ext_vector_type(4)))  _Float16 v4h;
typedef __attribute__((ext_vector_type(8)))  float    v8f;

#define WINS   15
#define BATCH  4096
#define NI     256
#define NA     64
#define DIM    320
#define HID    2048
#define DECAY  0.2f
#define THRESH 0.5f
#define HG     (HID / 16)   // 128 column-groups of 16

// ---------------------------------------------------------------------------
// Pre-pass: convert concat(state, action) -> packed f16 xh[B][T][D], once.
// Removes 256x-redundant f32->f16 conversion from the hot loop and halves
// the main kernel's global read traffic.
// ---------------------------------------------------------------------------
__global__ __launch_bounds__(256)
void snn_convert_kernel(const float* __restrict__ state,
                        const float* __restrict__ action,
                        _Float16* __restrict__ xh)
{
    const int idx = blockIdx.x * blockDim.x + threadIdx.x;   // one per 4 elems
    const int total4 = BATCH * WINS * DIM / 4;
    if (idx >= total4) return;
    const int e  = idx * 4;
    const int k  = e % DIM;          // 0..319, multiple of 4
    const int bt = e / DIM;          // fused (b*T + t)

    const float* __restrict__ src = (k < NI)
        ? (state  + (size_t)bt * NI + k)
        : (action + (size_t)bt * NA + (k - NI));
    const float4 v = *(const float4*)src;    // 16B aligned (k % 4 == 0)

    v4h o;
    o[0] = (_Float16)v.x; o[1] = (_Float16)v.y;
    o[2] = (_Float16)v.z; o[3] = (_Float16)v.w;
    *(v4h*)(xh + e) = o;                     // 8B store
}

// ---------------------------------------------------------------------------
// Fused kernel: per-wave 16x16 tile, WMMA GEMM over D=320 (10 x K=32),
// register-resident membrane recurrence over T=15, fused partial dot with Wo.
//
// Wave layout (wave32, v_wmma_f32_16x16x32_f16):
//   A (16x32 f16): lane m = lane&15 ; K half = (lane>=16 ? +8 : 0), halfs
//                  [0..7] = K kb..kb+7, [8..15] = K kb+16..kb+23
//   B (32x16 f16): lane n = lane&15 ; same K order as A: halfs [0..7] = K kb..kb+7,
//                  [8..15] = K kb+16..kb+23, kb = 8*(lane>=16)
//   C/D (16x16 f32): lane n = lane&15 ; acc[i] = element (m = 8*(lane>=16)+i, n)
// ---------------------------------------------------------------------------
__global__ __launch_bounds__(256)
void snn_fused_kernel(const _Float16* __restrict__ xh,
                      const float* __restrict__ W1, const float* __restrict__ b1,
                      const float* __restrict__ Wo1,
                      const float* __restrict__ W2, const float* __restrict__ b2,
                      const float* __restrict__ Wo2,
                      float* __restrict__ partials /* [2][BATCH][HG] */)
{
    const int lane    = threadIdx.x & 31;
    const int wave    = threadIdx.x >> 5;
    const int branch  = wave >> 2;       // 0 -> (W1,b1,Wo1), 1 -> (W2,b2,Wo2)
    const int nsub    = wave & 3;        // 16-col group within the 64-col tile

    const int b0 = blockIdx.x * 16;              // batch tile base
    const int hg = blockIdx.y * 4 + nsub;        // global 16-col group, 0..127
    const int n0 = hg * 16;

    const float* __restrict__ Wm   = branch ? W2  : W1;
    const float* __restrict__ bias = branch ? b2  : b1;
    const float* __restrict__ Wo   = branch ? Wo2 : Wo1;

    const int halfsel = lane >> 4;       // 0: lanes 0-15, 1: lanes 16-31
    const int ln      = lane & 15;       // row m for A loads / col n for B,C,D
    const int n       = n0 + ln;

    // ---- preload B fragments for all 10 K-chunks (time-invariant weights) ----
    v16h wfrag[10];
#pragma unroll
    for (int kc = 0; kc < 10; ++kc) {
        const int kb = kc * 32 + halfsel * 8;
#pragma unroll
        for (int v = 0; v < 8; ++v) {
            wfrag[kc][2*v]   = (_Float16)Wm[(size_t)(kb + 2*v + (v >= 4 ? 8 : 0)    ) * HID + n];
            wfrag[kc][2*v+1] = (_Float16)Wm[(size_t)(kb + 2*v + (v >= 4 ? 8 : 0) + 1) * HID + n];
        }
    }

    const float biasv = bias[n];
    const float wov   = Wo[n];

    v8f mreg = {}, sreg = {}, sumreg = {};

    // per-lane batch-row base pointer for A loads (row m = ln)
    const _Float16* __restrict__ xrow = xh + (size_t)(b0 + ln) * WINS * DIM;
    const int ka = halfsel * 8;          // half offset inside a 32-chunk

    for (int t = 0; t < WINS; ++t) {
        const _Float16* __restrict__ xp = xrow + t * DIM;

        v8f acc = {};
#pragma unroll
        for (int kc = 0; kc < 10; ++kc) {
            const _Float16* __restrict__ base = xp + kc * 32 + ka;
            // two 16B f16 vector loads, fragment-exact (offsets % 16B == 0)
            const v8h lo = *(const v8h*)(base);
            const v8h hi = *(const v8h*)(base + 16);
            const v16h afr = __builtin_shufflevector(
                lo, hi, 0, 1, 2, 3, 4, 5, 6, 7, 8, 9, 10, 11, 12, 13, 14, 15);
            acc = __builtin_amdgcn_wmma_f32_16x16x32_f16(
                false, afr, false, wfrag[kc], (short)0, acc, false, false);
        }

        // membrane recurrence on this tile's 8 (m,n) elements
#pragma unroll
        for (int i = 0; i < 8; ++i) {
            float mv = mreg[i] * (DECAY * (1.0f - sreg[i])) + acc[i] + biasv;
            float sv = (mv - THRESH) > 0.0f ? 1.0f : 0.0f;
            mreg[i]   = mv;
            sreg[i]   = sv;
            sumreg[i] += sv;
        }
    }

    // fused partial of (sum) @ Wo over this wave's 16 columns
    float p[8];
#pragma unroll
    for (int i = 0; i < 8; ++i) p[i] = sumreg[i] * wov;

    // butterfly reduce over the 16 lanes of each half (masks < 16 stay in-half)
#pragma unroll
    for (int i = 0; i < 8; ++i) {
#pragma unroll
        for (int off = 1; off < 16; off <<= 1)
            p[i] += __shfl_xor(p[i], off, 32);
    }

    if (ln == 0) {  // lane 0 -> rows b0..b0+7 ; lane 16 -> rows b0+8..b0+15
        const int mbase = b0 + halfsel * 8;
#pragma unroll
        for (int i = 0; i < 8; ++i)
            partials[((size_t)branch * BATCH + (mbase + i)) * HG + hg] = p[i];
    }
}

// ---------------------------------------------------------------------------
// Deterministic reduction of the 128 column-group partials per output element.
// ---------------------------------------------------------------------------
__global__ __launch_bounds__(256)
void snn_reduce_kernel(const float* __restrict__ partials,
                       const float* __restrict__ bo1,
                       const float* __restrict__ bo2,
                       float* __restrict__ out /* [2*BATCH]: out1 then out2 */)
{
    const int idx = blockIdx.x * blockDim.x + threadIdx.x;
    if (idx >= 2 * BATCH) return;
    const int branch = idx >> 12;  // idx/4096
    const float bo = branch ? bo2[0] : bo1[0];
    const float* __restrict__ p = partials + (size_t)idx * HG;
    float s = 0.0f;
#pragma unroll 8
    for (int i = 0; i < HG; ++i) s += p[i];
    out[idx] = s * (1.0f / WINS) + bo;
}

extern "C" void kernel_launch(void* const* d_in, const int* in_sizes, int n_in,
                              void* d_out, int out_size, void* d_ws, size_t ws_size,
                              hipStream_t stream) {
    (void)in_sizes; (void)n_in; (void)out_size; (void)ws_size;

    const float* state  = (const float*)d_in[0];
    const float* action = (const float*)d_in[1];
    const float* W1     = (const float*)d_in[2];
    const float* b1     = (const float*)d_in[3];
    const float* Wo1    = (const float*)d_in[4];
    const float* bo1    = (const float*)d_in[5];
    const float* W2     = (const float*)d_in[6];
    const float* b2     = (const float*)d_in[7];
    const float* Wo2    = (const float*)d_in[8];
    const float* bo2    = (const float*)d_in[9];

    // workspace layout: xh f16 [B][T][D] then partials f32 [2][B][HG]
    const size_t xh_bytes = (size_t)BATCH * WINS * DIM * sizeof(_Float16); // 39.3 MB
    _Float16* xh       = (_Float16*)d_ws;
    float*    partials = (float*)((char*)d_ws + xh_bytes);                 // 4 MB
    float*    out      = (float*)d_out;

    const int total4 = BATCH * WINS * DIM / 4;
    snn_convert_kernel<<<(total4 + 255) / 256, 256, 0, stream>>>(state, action, xh);

    dim3 grid(BATCH / 16, HID / 64);   // 256 x 32 workgroups
    snn_fused_kernel<<<grid, 256, 0, stream>>>(xh,
                                               W1, b1, Wo1,
                                               W2, b2, Wo2,
                                               partials);

    snn_reduce_kernel<<<(2 * BATCH + 255) / 256, 256, 0, stream>>>(
        partials, bo1, bo2, out);
}
